// MLoRACVR_3530463117981
// MI455X (gfx1250) — matrix operand order repaired and checked by reading, not compiled, not run on hardware
//
#include <hip/hip_runtime.h>
#include <hip/hip_bf16.h>
#include <stdint.h>

#define F_FEAT 8
#define VOCAB  100000
#define EDIM   32
#define HDIM   128
#define DDOM   19
#define RRANK  8
#define BATCH  32768
#define EPSV   1e-5f

typedef __attribute__((ext_vector_type(16))) __bf16          v16bf;
typedef __attribute__((ext_vector_type(16))) unsigned short  v16us;
typedef __attribute__((ext_vector_type(8)))  float           v8f;

struct U8x32 { uint4 lo, hi; };   // 32-byte LDS fragment (2 x ds_load_b128)

// native f32 -> bf16 (single v_cvt on gfx1250), RTE
__device__ __forceinline__ unsigned short f2bf(float f) {
    __bf16 h = (__bf16)f;
    return __builtin_bit_cast(unsigned short, h);
}
__device__ __forceinline__ float bf2f(unsigned short s) {
    union { uint32_t u; float f; } cv; cv.u = ((uint32_t)s) << 16;
    return cv.f;
}

// ---------------------------------------------------------------------------
// K0: LoRA term collapses to a per-(f,domain) bias:
//     bias[f,d,o] = sum_r (sum_i A[f,d,r,i]) * B[f,d,r,o]
// ---------------------------------------------------------------------------
__global__ void __launch_bounds__(128)
k_lora_bias(const float* __restrict__ A1, const float* __restrict__ B1,
            const float* __restrict__ A2, const float* __restrict__ B2,
            float* __restrict__ lora1, float* __restrict__ lora2) {
    const int fd = blockIdx.x;           // f*DDOM + d
    const int o  = threadIdx.x;          // 0..127
    __shared__ float sA1[RRANK], sA2[RRANK];
    if (o < RRANK) {
        const float* a1 = A1 + (size_t)fd * (RRANK * EDIM) + o * EDIM;
        float s = 0.f;
        for (int i = 0; i < EDIM; ++i) s += a1[i];
        sA1[o] = s;
        const float* a2 = A2 + (size_t)fd * (RRANK * HDIM) + o * HDIM;
        float s2 = 0.f;
        for (int i = 0; i < HDIM; ++i) s2 += a2[i];
        sA2[o] = s2;
    }
    __syncthreads();
    const float* b1 = B1 + (size_t)fd * (RRANK * HDIM);
    const float* b2 = B2 + (size_t)fd * (RRANK * HDIM);
    float acc1 = 0.f, acc2 = 0.f;
#pragma unroll
    for (int r = 0; r < RRANK; ++r) {
        acc1 += sA1[r] * b1[r * HDIM + o];
        acc2 += sA2[r] * b2[r * HDIM + o];
    }
    lora1[(size_t)fd * HDIM + o] = acc1;
    lora2[(size_t)fd * HDIM + o] = acc2;
}

// ---------------------------------------------------------------------------
// K1: layer1 GEMM per feature f: [128-row tile x E=32] x [32 x 128] via one
// v_wmma_f32_16x16x32_bf16 per 16x16 output tile (K == E == 32).
// W1 is staged in LDS PRE-SWIZZLED into B-fragment order so each lane pulls
// its whole fragment with two ds_load_b128.
// ---------------------------------------------------------------------------
__global__ void __launch_bounds__(256)
k_layer1(const int* __restrict__ x_str, const int* __restrict__ dom_ids,
         const float* __restrict__ emb, const float* __restrict__ W1,
         const float* __restrict__ lora1,
         unsigned short* __restrict__ h1,
         float* __restrict__ gsum, float* __restrict__ gsq) {
    const int tile = blockIdx.x & 255;           // 256 tiles of 128 rows
    const int f    = blockIdx.x >> 8;
    const int tid  = threadIdx.x;
    const int wave = tid >> 5;                   // 0..7
    const int lane = tid & 31;
    const int rowBlockBase = tile * 128;

    // W1sw: fragment-order [n (8)][lane (32)][t (16)] bf16 = 8 KB
    __shared__ unsigned short W1sw[8 * 32 * 16];
    __shared__ float lora1s[DDOM * HDIM];              // 9.5 KB
    __shared__ int   doms[128];
    __shared__ float ssum[HDIM], ssq[HDIM];
    __shared__ unsigned short outs[8][16 * HDIM];      // 32 KB staging

    // stage W1[f] bf16, swizzled: element (k,col) -> [(col>>4)][ ((k>>3)&1)*16 + (col&15) ][ (k&7) | ((k>>4)<<3) ]
    const float* W1f = W1 + (size_t)f * EDIM * HDIM;
    for (int i = tid; i < EDIM * HDIM; i += 256) {
        int k = i >> 7, col = i & 127;
        int n = col >> 4;
        int lanei = (((k >> 3) & 1) << 4) + (col & 15);
        int t = (k & 7) | ((k >> 4) << 3);
        W1sw[(n * 32 + lanei) * 16 + t] = f2bf(W1f[i]);
    }
    const float* l1f = lora1 + (size_t)f * DDOM * HDIM;
    for (int i = tid; i < DDOM * HDIM; i += 256) lora1s[i] = l1f[i];
    if (tid < 128) doms[tid] = dom_ids[rowBlockBase + tid];
    if (tid < HDIM) { ssum[tid] = 0.f; ssq[tid] = 0.f; }
    __syncthreads();

    // ---- A fragment: 16x32 bf16. lanes 0-15: K 0-7,16-23; lanes 16-31: K 8-15,24-31
    const int half = lane >> 4;
    const int lr   = lane & 15;
    const int rowWaveBase = rowBlockBase + wave * 16;
    const int r = rowWaveBase + lr;
    int xi = x_str[(size_t)r * F_FEAT + f];
    xi = min(max(xi, 0), VOCAB - 1);
    const float* erow = emb + ((size_t)f * VOCAB + (size_t)xi) * EDIM;
    const int koff = half * 8;
    float4 p0 = *(const float4*)(erow + koff);
    float4 p1 = *(const float4*)(erow + koff + 4);
    float4 p2 = *(const float4*)(erow + koff + 16);
    float4 p3 = *(const float4*)(erow + koff + 20);
    v16us aus;
    aus[0]  = f2bf(p0.x); aus[1]  = f2bf(p0.y); aus[2]  = f2bf(p0.z); aus[3]  = f2bf(p0.w);
    aus[4]  = f2bf(p1.x); aus[5]  = f2bf(p1.y); aus[6]  = f2bf(p1.z); aus[7]  = f2bf(p1.w);
    aus[8]  = f2bf(p2.x); aus[9]  = f2bf(p2.y); aus[10] = f2bf(p2.z); aus[11] = f2bf(p2.w);
    aus[12] = f2bf(p3.x); aus[13] = f2bf(p3.y); aus[14] = f2bf(p3.z); aus[15] = f2bf(p3.w);
    v16bf a = __builtin_bit_cast(v16bf, aus);

    for (int n = 0; n < 8; ++n) {
        const int col = n * 16 + lr;
        U8x32 w = *(const U8x32*)&W1sw[(n * 32 + lane) * 16];   // 2x ds_load_b128
        v16bf b = __builtin_bit_cast(v16bf, w);
        v8f c = {};
        c = __builtin_amdgcn_wmma_f32_16x16x32_bf16(false, a, false, b,
                                                    (short)0, c, false, false);
        float psum = 0.f, psq = 0.f;
#pragma unroll
        for (int v = 0; v < 8; ++v) {
            const int m = half * 8 + v;        // C layout: VGPR v -> row v / v+8
            float hv = c[v] + lora1s[doms[wave * 16 + m] * HDIM + col];
            psum += hv; psq += hv * hv;
            outs[wave][m * HDIM + col] = f2bf(hv);
        }
        atomicAdd(&ssum[col], psum);
        atomicAdd(&ssq[col],  psq);
    }
    __syncthreads();

    // coalesced b128 stores of the wave's 16x128 bf16 tile
#pragma unroll
    for (int p = 0; p < 8; ++p) {
        int flat = p * 256 + lane * 8;         // ushort index in tile
        int row  = flat >> 7;
        int cu   = flat & 127;
        uint4 q = *(const uint4*)&outs[wave][row * HDIM + cu];
        unsigned short* dst = h1 + ((size_t)f * BATCH + rowWaveBase + row) * HDIM + cu;
        *(uint4*)dst = q;
    }
    if (tid < HDIM) {
        atomicAdd(&gsum[f * HDIM + tid], ssum[tid]);
        atomicAdd(&gsq[f * HDIM + tid],  ssq[tid]);
    }
}

// ---------------------------------------------------------------------------
// K2: fold BN stats into affine scale/shift per (f,o)
// ---------------------------------------------------------------------------
__global__ void k_bnstats(const float* __restrict__ gsum, const float* __restrict__ gsq,
                          const float* __restrict__ g, const float* __restrict__ be,
                          float* __restrict__ sc, float* __restrict__ sh) {
    int i = blockIdx.x * blockDim.x + threadIdx.x;
    if (i >= F_FEAT * HDIM) return;
    float mean = gsum[i] * (1.f / (float)BATCH);
    float var  = gsq[i]  * (1.f / (float)BATCH) - mean * mean;
    float inv  = rsqrtf(var + EPSV);
    float s = g[i] * inv;
    sc[i] = s;
    sh[i] = be[i] - mean * s;
}

// ---------------------------------------------------------------------------
// K3: layer2: BN1+ReLU applied on the fly to h1, then [64 x 128] x [128 x 128]
// per feature via 4 chained v_wmma_f32_16x16x32_bf16 (K=128).
// W2 staged pre-swizzled: [n (8)][kb (4)][lane (32)][t (16)] bf16 = 32 KB.
// ---------------------------------------------------------------------------
__global__ void __launch_bounds__(128)
k_layer2(const int* __restrict__ dom_ids, const unsigned short* __restrict__ h1,
         const float* __restrict__ W2, const float* __restrict__ lora2,
         const float* __restrict__ sc1, const float* __restrict__ sh1,
         unsigned short* __restrict__ h2,
         float* __restrict__ gsum, float* __restrict__ gsq) {
    const int tile = blockIdx.x & 511;           // 512 tiles of 64 rows
    const int f    = blockIdx.x >> 9;
    const int tid  = threadIdx.x;
    const int wave = tid >> 5;                   // 0..3
    const int lane = tid & 31;
    const int rowBlockBase = tile * 64;

    __shared__ unsigned short W2sw[HDIM * HDIM];       // 32 KB, fragment order
    __shared__ float lora2s[DDOM * HDIM];              // 9.5 KB
    __shared__ int   doms[64];
    __shared__ float sc1s[HDIM], sh1s[HDIM];
    __shared__ float ssum[HDIM], ssq[HDIM];
    __shared__ unsigned short outs[4][16 * HDIM];      // 16 KB

    // stage W2[f] bf16 swizzled: (k,col) -> [(col>>4)*4 + (k>>5)][ ((k>>3)&1)*16 + (col&15) ][ (k&7) | (((k&31)>>4)<<3) ]
    const float* W2f = W2 + (size_t)f * HDIM * HDIM;
    for (int i = tid; i < HDIM * HDIM; i += 128) {
        int k = i >> 7, col = i & 127;
        int n = col >> 4;
        int kb = k >> 5;
        int krem = k & 31;
        int lanei = (((krem >> 3) & 1) << 4) + (col & 15);
        int t = (krem & 7) | ((krem >> 4) << 3);
        W2sw[((n * 4 + kb) * 32 + lanei) * 16 + t] = f2bf(W2f[i]);
    }
    const float* l2f = lora2 + (size_t)f * DDOM * HDIM;
    for (int i = tid; i < DDOM * HDIM; i += 128) lora2s[i] = l2f[i];
    if (tid < 64) doms[tid] = dom_ids[rowBlockBase + tid];
    if (tid < HDIM) {
        sc1s[tid] = sc1[f * HDIM + tid];
        sh1s[tid] = sh1[f * HDIM + tid];
        ssum[tid] = 0.f; ssq[tid] = 0.f;
    }
    __syncthreads();

    const int half = lane >> 4;
    const int lr   = lane & 15;
    const int rowWaveBase = rowBlockBase + wave * 16;
    const int r = rowWaveBase + lr;
    const unsigned short* hrow = h1 + ((size_t)f * BATCH + r) * HDIM;
    const int koff = half * 8;

    // Build 4 A fragments (K = 128 as 4 slices of 32), BN+ReLU fused.
    v16bf afr[4];
#pragma unroll
    for (int kb = 0; kb < 4; ++kb) {
        const int k0 = kb * 32 + koff;
        uint4 q0 = *(const uint4*)(hrow + k0);
        uint4 q1 = *(const uint4*)(hrow + k0 + 16);
        const unsigned short* u0 = (const unsigned short*)&q0;
        const unsigned short* u1 = (const unsigned short*)&q1;
        v16us aus;
#pragma unroll
        for (int t = 0; t < 8; ++t) {
            int ka = k0 + t;
            float x = fmaxf(bf2f(u0[t]) * sc1s[ka] + sh1s[ka], 0.f);
            aus[t] = f2bf(x);
            int kbk = k0 + 16 + t;
            float y = fmaxf(bf2f(u1[t]) * sc1s[kbk] + sh1s[kbk], 0.f);
            aus[8 + t] = f2bf(y);
        }
        afr[kb] = __builtin_bit_cast(v16bf, aus);
    }

    for (int n = 0; n < 8; ++n) {
        const int col = n * 16 + lr;
        v8f c = {};
#pragma unroll
        for (int kb = 0; kb < 4; ++kb) {
            U8x32 w = *(const U8x32*)&W2sw[((n * 4 + kb) * 32 + lane) * 16]; // 2x ds_load_b128
            v16bf b = __builtin_bit_cast(v16bf, w);
            c = __builtin_amdgcn_wmma_f32_16x16x32_bf16(false, afr[kb], false, b,
                                                        (short)0, c, false, false);
        }
        float psum = 0.f, psq = 0.f;
#pragma unroll
        for (int v = 0; v < 8; ++v) {
            const int m = half * 8 + v;
            float hv = c[v] + lora2s[doms[wave * 16 + m] * HDIM + col];
            psum += hv; psq += hv * hv;
            outs[wave][m * HDIM + col] = f2bf(hv);
        }
        atomicAdd(&ssum[col], psum);
        atomicAdd(&ssq[col],  psq);
    }
    __syncthreads();
#pragma unroll
    for (int p = 0; p < 8; ++p) {
        int flat = p * 256 + lane * 8;
        int row  = flat >> 7;
        int cu   = flat & 127;
        uint4 q = *(const uint4*)&outs[wave][row * HDIM + cu];
        unsigned short* dst = h2 + ((size_t)f * BATCH + rowWaveBase + row) * HDIM + cu;
        *(uint4*)dst = q;
    }
    if (tid < HDIM) {
        atomicAdd(&gsum[f * HDIM + tid], ssum[tid]);
        atomicAdd(&gsq[f * HDIM + tid],  ssq[tid]);
    }
}

// ---------------------------------------------------------------------------
// K4: BN2 + ReLU + sum over f + head dot + sigmoid; one wave per batch row.
// ---------------------------------------------------------------------------
__global__ void __launch_bounds__(256)
k_head(const unsigned short* __restrict__ h2,
       const float* __restrict__ sc2, const float* __restrict__ sh2,
       const float* __restrict__ headW, const float* __restrict__ headb,
       float* __restrict__ out) {
    const int tid  = threadIdx.x;
    const int wave = tid >> 5;
    const int lane = tid & 31;
    const int b = blockIdx.x * 8 + wave;
    __shared__ float sc2s[F_FEAT * HDIM], sh2s[F_FEAT * HDIM];
    for (int i = tid; i < F_FEAT * HDIM; i += 256) { sc2s[i] = sc2[i]; sh2s[i] = sh2[i]; }
    __syncthreads();

    const int c0 = lane * 4;
    float4 hw = *(const float4*)(headW + c0);
    const float* hwp = (const float*)&hw;
    float acc = 0.f;
#pragma unroll
    for (int f = 0; f < F_FEAT; ++f) {
        const unsigned short* hp = h2 + ((size_t)f * BATCH + b) * HDIM + c0;
        uint2 q = *(const uint2*)hp;
        const unsigned short* u = (const unsigned short*)&q;
#pragma unroll
        for (int j = 0; j < 4; ++j) {
            int o = c0 + j;
            float x = fmaxf(bf2f(u[j]) * sc2s[f * HDIM + o] + sh2s[f * HDIM + o], 0.f);
            acc += x * hwp[j];
        }
    }
#pragma unroll
    for (int m = 16; m >= 1; m >>= 1) acc += __shfl_xor(acc, m, 32);
    if (lane == 0) out[b] = 1.f / (1.f + __expf(-(acc + headb[0])));
}

// ---------------------------------------------------------------------------
extern "C" void kernel_launch(void* const* d_in, const int* in_sizes, int n_in,
                              void* d_out, int out_size, void* d_ws, size_t ws_size,
                              hipStream_t stream) {
    const int*   x_str = (const int*)  d_in[0];
    const int*   dom   = (const int*)  d_in[1];
    const float* emb   = (const float*)d_in[2];
    const float* W1    = (const float*)d_in[3];
    const float* A1    = (const float*)d_in[4];
    const float* B1    = (const float*)d_in[5];
    const float* g1    = (const float*)d_in[6];
    const float* be1   = (const float*)d_in[7];
    const float* W2    = (const float*)d_in[8];
    const float* A2    = (const float*)d_in[9];
    const float* B2    = (const float*)d_in[10];
    const float* g2    = (const float*)d_in[11];
    const float* be2   = (const float*)d_in[12];
    const float* headW = (const float*)d_in[13];
    const float* headb = (const float*)d_in[14];
    float* out = (float*)d_out;

    char* ws = (char*)d_ws;
    float* lora1 = (float*)(ws + 0);          //  8*19*128 f32
    float* lora2 = (float*)(ws + 77824);
    float* sum1  = (float*)(ws + 155648);     // 1024 f32 each
    float* sq1   = (float*)(ws + 159744);
    float* sum2  = (float*)(ws + 163840);
    float* sq2   = (float*)(ws + 167936);
    float* sc1   = (float*)(ws + 172032);
    float* sh1   = (float*)(ws + 176128);
    float* sc2   = (float*)(ws + 180224);
    float* sh2   = (float*)(ws + 184320);
    unsigned short* h1 = (unsigned short*)(ws + 188416);                 // 64 MB bf16
    unsigned short* h2 = (unsigned short*)(ws + 188416 +
                           (size_t)F_FEAT * BATCH * HDIM * 2);           // 64 MB bf16

    // zero the BN stat accumulators (graph-capture legal)
    hipMemsetAsync(ws + 155648, 0, 4 * 4096, stream);

    k_lora_bias<<<F_FEAT * DDOM, 128, 0, stream>>>(A1, B1, A2, B2, lora1, lora2);
    k_layer1<<<F_FEAT * (BATCH / 128), 256, 0, stream>>>(x_str, dom, emb, W1, lora1,
                                                         h1, sum1, sq1);
    k_bnstats<<<4, 256, 0, stream>>>(sum1, sq1, g1, be1, sc1, sh1);
    k_layer2<<<F_FEAT * (BATCH / 64), 128, 0, stream>>>(dom, h1, W2, lora2, sc1, sh1,
                                                        h2, sum2, sq2);
    k_bnstats<<<4, 256, 0, stream>>>(sum2, sq2, g2, be2, sc2, sh2);
    k_head<<<BATCH / 8, 256, 0, stream>>>(h2, sc2, sh2, headW, headb, out);
}
